// model0_33097017983659
// MI455X (gfx1250) — matrix operand order repaired and checked by reading, not compiled, not run on hardware
//
#include <hip/hip_runtime.h>
#include <hip/hip_bf16.h>
#include <math.h>

// HMM forward log-likelihood for MI455X (gfx1250, wave32).
// Pass 0: precompute f16 emission-weight WMMA B-fragments + bias.
// Pass 1: emission GEMM via v_wmma_f32_16x16x32_f16; seq streamed with
//         non-temporal float4 loads; row-max via DPP butterfly; store
//         pe=exp(emit-rowmax) (f16) + rowmax (f32) -- ~36MB, L2-resident.
// Pass 2: linear-space forward recursion via v_wmma_f32_16x16x4_f32 with a
//         permuted hidden index so B-operands are plain register copies;
//         renormalization every 4 steps, fast rcp/log off the critical path.

typedef __attribute__((ext_vector_type(16))) _Float16 v16h;
typedef __attribute__((ext_vector_type(8)))  float    v8f;
typedef __attribute__((ext_vector_type(4)))  float    v4f;
typedef __attribute__((ext_vector_type(2)))  float    v2f;

#define BB 2048
#define TT 512
#define DD 88
#define HH 16

// workspace layout (bytes); requires ~37.8 MB
#define WT_OFF   0                        // 3*32*16 halves = 3072 B (B-fragment layout)
#define BIAS_OFF 3072                     // 16 f32
#define EMAX_OFF 4096                     // BB*TT f32 = 4 MB, [t][i]
#define PE_OFF   (4096 + BB*TT*4)         // TT*HH*BB f16 = 32 MB, [t][h][i]

// fmax with a DPP-permuted copy of v (pure VALU cross-lane)
#define DPP_FMAX(v, ctrl)                                                     \
  v = fmaxf(v, __int_as_float(__builtin_amdgcn_update_dpp(                    \
          __float_as_int(v), __float_as_int(v), (ctrl), 0xf, 0xf, true)))

__device__ __forceinline__ float swz_xor16(float x) {
  // ds_swizzle group-of-32: and=0x1f, or=0, xor=0x10  -> lane ^ 16
  return __int_as_float(__builtin_amdgcn_ds_swizzle(__float_as_int(x), 0x401F));
}

// ---------------- Pass 0: weight fragments + bias -------------------------
__global__ __launch_bounds__(256) void hmm_pass0(const float* __restrict__ py,
                                                 char* __restrict__ ws) {
  _Float16* wt   = (_Float16*)(ws + WT_OFF);
  float*    bias = (float*)(ws + BIAS_OFF);
  int tid = threadIdx.x;
  // wt fragment: idx = s*512 + lane*16 + e ; element (K=(lane<16?0:16)+e, N=lane&15)
  for (int idx = tid; idx < 3 * 32 * 16; idx += 256) {
    int s = idx >> 9;
    int l = (idx >> 4) & 31;
    int e = idx & 15;
    int n = l & 15;
    int k = (l < 16 ? 0 : 16) + e;
    int d = s * 32 + k;
    float v = 0.0f;
    if (d < DD) {
      float p = py[n * DD + d];
      v = logf(p) - log1pf(-p);
    }
    wt[idx] = (_Float16)v;
  }
  if (tid < HH) {
    float s = 0.0f;
    for (int d = 0; d < DD; ++d) s += log1pf(-py[tid * DD + d]);
    bias[tid] = s;
  }
}

// ---------------- Pass 1: emission GEMM (WMMA f16) ------------------------
// one wave per 16-row tile of the flattened [B*T, D] matrix (rows share one b)
__global__ __launch_bounds__(128) void hmm_pass1(const float* __restrict__ seq,
                                                 const long long* __restrict__ mb,
                                                 char* __restrict__ ws) {
  const int lane = threadIdx.x & 31;
  const int hf   = lane >> 4;      // half of the wave (0/1)
  const int m    = lane & 15;
  const int tile = blockIdx.x * 4 + (threadIdx.x >> 5);
  const int row0 = tile << 4;
  const int i    = row0 >> 9;          // row0 / TT
  const int tb   = row0 & (TT - 1);    // row0 % TT

  const _Float16* wt     = (const _Float16*)(ws + WT_OFF);
  const float*    bias   = (const float*)(ws + BIAS_OFF);
  float*          emaxWS = (float*)(ws + EMAX_OFF);
  _Float16*       peWS   = (_Float16*)(ws + PE_OFF);

  const float* srow = seq + (size_t)mb[i] * (size_t)(TT * DD) + (size_t)(tb + m) * DD;

  // A fragment: elements e=0..7 <-> K = 16*hf + 0..7 (contiguous run),
  //             elements e=8..15 <-> K = 16*hf + 8 + 0..7 (contiguous run)
  v16h a[3], b[3];
#pragma unroll
  for (int s = 0; s < 3; ++s) {
    b[s] = *(const v16h*)(wt + (s * 32 + lane) * 16);
#pragma unroll
    for (int run = 0; run < 2; ++run) {
      const int d0 = s * 32 + 16 * hf + run * 8;     // multiple of 8 -> 32B aligned
      v4f lo = {0.f, 0.f, 0.f, 0.f}, hi = {0.f, 0.f, 0.f, 0.f};
      if (d0 < DD) {
        lo = __builtin_nontemporal_load((const v4f*)(srow + d0));
        hi = __builtin_nontemporal_load((const v4f*)(srow + d0 + 4));
      }
#pragma unroll
      for (int j = 0; j < 4; ++j) {
        a[s][run * 8 + j]     = (_Float16)lo[j];
        a[s][run * 8 + 4 + j] = (_Float16)hi[j];
      }
    }
  }

  v8f c = {0.f, 0.f, 0.f, 0.f, 0.f, 0.f, 0.f, 0.f};
  c = __builtin_amdgcn_wmma_f32_16x16x32_f16(false, a[0], false, b[0], (short)0, c, false, false);
  c = __builtin_amdgcn_wmma_f32_16x16x32_f16(false, a[1], false, b[1], (short)0, c, false, false);
  c = __builtin_amdgcn_wmma_f32_16x16x32_f16(false, a[2], false, b[2], (short)0, c, false, false);

  const float bn = bias[m];   // column h = lane&15
#pragma unroll
  for (int r = 0; r < 8; ++r) {
    float ev = c[r] + bn;     // emit for row t=tb+r+8*hf, col h=m
    float mx = ev;            // max across the 16 lanes of this half, via DPP butterfly
    DPP_FMAX(mx, 0xB1);       // quad_perm(1,0,3,2)  : xor 1
    DPP_FMAX(mx, 0x4E);       // quad_perm(2,3,0,1)  : xor 2
    DPP_FMAX(mx, 0x141);      // row_half_mirror     : xor 7
    DPP_FMAX(mx, 0x140);      // row_mirror          : xor 15
    const int t = tb + r + 8 * hf;
    peWS[((size_t)t * HH + m) * BB + i] = (_Float16)__expf(ev - mx);
    if (m == 0) emaxWS[(size_t)t * BB + i] = mx;
  }
}

// ---------------- Pass 2: forward recursion (WMMA f32 16x16x4) ------------
// hidden index permuted: state slot (reg r, half hf) holds
//   alpha[m = lane&15][h = 4*(r>>1) + (r&1) + 2*hf]
// which is exactly the K element the next step's B-operand chunk q=r>>1,
// j=r&1 needs -> B fragments are plain register copies.
__global__ __launch_bounds__(32) void hmm_pass2(const float* __restrict__ px,
                                                const float* __restrict__ scalep,
                                                const int* __restrict__ lengths,
                                                const long long* __restrict__ mb,
                                                const unsigned char* __restrict__ maskp,
                                                const char* __restrict__ ws,
                                                float* __restrict__ out) {
  const int lane  = threadIdx.x & 31;
  const int hf    = lane >> 4;
  const int m     = lane & 15;
  const int ibase = blockIdx.x << 4;
  const int i     = ibase + m;

  const float*    emaxWS = (const float*)(ws + EMAX_OFF);
  const _Float16* peWS   = (const _Float16*)(ws + PE_OFF);

  const int len = lengths[(int)mb[i]];
  // output-row permutation for this lane's A-fragment row (M = m)
  const int pm = 4 * ((m & 7) >> 1) + (m & 1) + 2 * (m >> 3);

  // constant A^T fragment: element (q, j) at this lane = A[k = 4q+j+2hf][h' = perm(m)]
  v2f af[4];
#pragma unroll
  for (int q = 0; q < 4; ++q)
#pragma unroll
    for (int j = 0; j < 2; ++j)
      af[q][j] = px[(4 * q + j + 2 * hf) * HH + pm];

  v8f al = {0.f, 0.f, 0.f, 0.f, 0.f, 0.f, 0.f, 0.f};
  al[0] = (hf == 0) ? 1.0f : 0.0f;   // slot (0,0) is h=0: delta at state 0
  float cacc = 0.0f;

  for (int t0 = 0; t0 < TT; t0 += 4) {
#pragma unroll
    for (int k = 0; k < 4; ++k) {
      const int t = t0 + k;
      // B fragments are direct register pairs of the permuted state
      v2f bf[4];
#pragma unroll
      for (int q = 0; q < 4; ++q) { bf[q][0] = al[2 * q]; bf[q][1] = al[2 * q + 1]; }

      // two independent 2-deep accumulation chains
      v8f z = {0.f, 0.f, 0.f, 0.f, 0.f, 0.f, 0.f, 0.f};
      v8f nxA = __builtin_amdgcn_wmma_f32_16x16x4_f32(false, af[0], false, bf[0], (short)0, z,   false, false);
      v8f nxB = __builtin_amdgcn_wmma_f32_16x16x4_f32(false, af[1], false, bf[1], (short)0, z,   false, false);
      nxA     = __builtin_amdgcn_wmma_f32_16x16x4_f32(false, af[2], false, bf[2], (short)0, nxA, false, false);
      nxB     = __builtin_amdgcn_wmma_f32_16x16x4_f32(false, af[3], false, bf[3], (short)0, nxB, false, false);

      // multiply by pe[t][h][i] with permuted h (coalesced in i across lanes)
      const _Float16* pt = peWS + (size_t)t * HH * BB + (size_t)(2 * hf) * BB + i;
      const bool valid = (t < len);
#pragma unroll
      for (int r = 0; r < 8; ++r) {
        const int hbase = 4 * (r >> 1) + (r & 1);   // + 2*hf folded into pt
        const float v = (nxA[r] + nxB[r]) * (float)pt[(size_t)hbase * BB];
        al[r] = valid ? v : al[r];
      }
      cacc += valid ? emaxWS[(size_t)t * BB + i] : 0.0f;

      if (t + 1 < TT) __builtin_prefetch(pt + (size_t)HH * BB, 0, 0);
    }
    // periodic renormalization (scale-invariant: commutes with masking)
    float s = ((al[0] + al[1]) + (al[2] + al[3])) + ((al[4] + al[5]) + (al[6] + al[7]));
    s += swz_xor16(s);
    const float inv = __builtin_amdgcn_rcpf(s);
#pragma unroll
    for (int r = 0; r < 8; ++r) al[r] *= inv;
    cacc += __logf(s);
  }

  float fs = ((al[0] + al[1]) + (al[2] + al[3])) + ((al[4] + al[5]) + (al[6] + al[7]));
  fs += swz_xor16(fs);
  const float ll = cacc + __logf(fs);

  if (hf == 0) {
    out[i] = maskp[i] ? (scalep[0] * ll) : 0.0f;
  }
}

// ---------------- host launch ---------------------------------------------
extern "C" void kernel_launch(void* const* d_in, const int* in_sizes, int n_in,
                              void* d_out, int out_size, void* d_ws, size_t ws_size,
                              hipStream_t stream) {
  (void)in_sizes; (void)n_in; (void)out_size; (void)ws_size;
  const float*         seq     = (const float*)d_in[0];        // [B,T,D] f32
  const float*         px      = (const float*)d_in[1];        // [H,H] f32
  const float*         py      = (const float*)d_in[2];        // [H,D] f32
  const float*         scalep  = (const float*)d_in[3];        // [1] f32
  const int*           lengths = (const int*)d_in[4];          // [B] i32
  const long long*     mb      = (const long long*)d_in[5];    // [B] i64
  const unsigned char* maskp   = (const unsigned char*)d_in[6];// [B] bool
  float* out = (float*)d_out;
  char*  ws  = (char*)d_ws;

  hmm_pass0<<<1, 256, 0, stream>>>(py, ws);
  hmm_pass1<<<(BB * TT / 16) / 4, 128, 0, stream>>>(seq, mb, ws);
  hmm_pass2<<<BB / 16, 32, 0, stream>>>(px, scalep, lengths, mb, maskp, ws, out);
}
